// PPA_67843303408299
// MI455X (gfx1250) — matrix operand order repaired and checked by reading, not compiled, not run on hardware
//
#include <hip/hip_runtime.h>
#include <hip/hip_bf16.h>

typedef __attribute__((ext_vector_type(16))) _Float16 v16h;
typedef __attribute__((ext_vector_type(8)))  _Float16 v8h;
typedef __attribute__((ext_vector_type(8)))  float    v8f;

#define B_     8
#define CIN    320
#define COUT   256
#define P_     16384          // H*W
#define NCLS   21
#define PIXB   64             // pixels per tile
#define NTIL   4              // N-tiles per block (PIXB/16)
#define KT     10             // 320 / 32
#define CPAD   (CIN + 8)      // 328, LDS pad for f16 stage
#define SOPAD  (PIXB + 1)     // 65, stride-65 -> conflict-free columns
#define NTOT   131072.0f      // B*H*W for BatchNorm

// ---- workspace layout (float offsets) ----
#define WS_CHSUM   0
#define WS_CHSQ    256
#define WS_SCALE   512
#define WS_SHIFT   768
#define WS_CLS_SS  1024                     // 8*21*256 = 43008
#define WS_CLS_QS  (WS_CLS_SS + 43008)      // 44032
#define WS_CLS_ST  (WS_CLS_QS + 43008)      // 87040
#define WS_CLS_QT  (WS_CLS_ST + 43008)      // 130048
#define WS_CNT     (WS_CLS_QT + 43008)      // 173056 (168 entries)
#define WS_MSE     (WS_CNT + 168)           // 173224
#define WS_ZEROCNT 173232                   // zero everything below Wh
#define WS_WH      173232                   // f16 W_proj lives here (16B aligned)

// ---------------- shared GEMM tile body ----------------
// Computes 2 M-tiles x NTIL N-tiles of W(256x320) * S(320xPIXB) with
// v_wmma_f32_16x16x32_f16.  A from global f16 W (row-major [o][c]),
// B from LDS f16 stage [p][CPAD].
#define GEMM_TILE_BODY(ACC)                                                     \
  {                                                                             \
    const int mrow0 = (wave * 2 + 0) * 16 + (lane & 15);                        \
    const int mrow1 = (wave * 2 + 1) * 16 + (lane & 15);                        \
    const int koff  = (lane >> 4) * 8;                                          \
    _Pragma("unroll")                                                           \
    for (int kt = 0; kt < KT; ++kt) {                                           \
      const int k0 = kt * 32;                                                   \
      const _Float16* wp0 = Wh + (size_t)mrow0 * CIN + k0 + koff;               \
      const _Float16* wp1 = Wh + (size_t)mrow1 * CIN + k0 + koff;               \
      v8h a0l = *(const v8h*)(wp0);  v8h a0h = *(const v8h*)(wp0 + 16);         \
      v8h a1l = *(const v8h*)(wp1);  v8h a1h = *(const v8h*)(wp1 + 16);         \
      v16h A0 = __builtin_shufflevector(a0l, a0h,                               \
                 0,1,2,3,4,5,6,7,8,9,10,11,12,13,14,15);                        \
      v16h A1 = __builtin_shufflevector(a1l, a1h,                               \
                 0,1,2,3,4,5,6,7,8,9,10,11,12,13,14,15);                        \
      const int kb = k0 + (lane >> 4) * 8;                                      \
      _Pragma("unroll")                                                         \
      for (int nt = 0; nt < NTIL; ++nt) {                                       \
        const int pix = nt * 16 + (lane & 15);                                  \
        const _Float16* bp = &sStage[pix * CPAD + kb];                          \
        v8h bl = *(const v8h*)(bp);                                             \
        v8h bh = *(const v8h*)(bp + 16);                                        \
        v16h Bm = __builtin_shufflevector(bl, bh,                               \
                   0,1,2,3,4,5,6,7,8,9,10,11,12,13,14,15);                      \
        ACC[0][nt] = __builtin_amdgcn_wmma_f32_16x16x32_f16(                    \
            false, A0, false, Bm, (short)0, ACC[0][nt], false, false);          \
        ACC[1][nt] = __builtin_amdgcn_wmma_f32_16x16x32_f16(                    \
            false, A1, false, Bm, (short)0, ACC[1][nt], false, false);          \
      }                                                                         \
    }                                                                           \
  }

// Stage one [CIN x PIXB] fp32 s-tile into LDS as f16 laid out [p][c].
#define STAGE_S_TILE()                                                          \
  {                                                                             \
    const float* sB = s_feat + (size_t)b * CIN * P_ + p0;                       \
    _Pragma("unroll 4")                                                         \
    for (int i = 0; i < (CIN * PIXB) / (256 * 4); ++i) {                        \
      int g = tid + i * 256;                                                    \
      int c = g >> 4;                                                           \
      int p = (g & 15) * 4;                                                     \
      const float4 v = *(const float4*)(sB + (size_t)c * P_ + p);               \
      sStage[(p + 0) * CPAD + c] = (_Float16)v.x;                               \
      sStage[(p + 1) * CPAD + c] = (_Float16)v.y;                               \
      sStage[(p + 2) * CPAD + c] = (_Float16)v.z;                               \
      sStage[(p + 3) * CPAD + c] = (_Float16)v.w;                               \
    }                                                                           \
  }

// ---------------- K0: zero accumulators ----------------
__global__ void k_zero(float* ws, int n) {
  int i = blockIdx.x * 256 + threadIdx.x;
  if (i < n) ws[i] = 0.0f;
}

// ---------------- Kw: W_proj fp32 -> f16 ----------------
__global__ void k_wcvt(const float* __restrict__ W, _Float16* __restrict__ Wh) {
  int i = blockIdx.x * 256 + threadIdx.x;   // 81920 total
  Wh[i] = (_Float16)W[i];
}

// ---------------- K1: WMMA GEMM pass 1 -> per-channel sum / sumsq ----------------
__global__ __launch_bounds__(256) void k1_bnstats(
    const float* __restrict__ s_feat, const _Float16* __restrict__ Wh,
    float* __restrict__ gChSum, float* __restrict__ gChSq) {
  __shared__ _Float16 sStage[PIXB * CPAD];
  __shared__ float chSum[COUT];
  __shared__ float chSq[COUT];
  const int tid = threadIdx.x;
  const int wave = tid >> 5, lane = tid & 31;
  const int tile = blockIdx.x;           // 2048 tiles
  const int b = tile >> 8;               // 256 tiles per batch
  const int p0 = (tile & 255) * PIXB;

  chSum[tid] = 0.0f;
  chSq[tid] = 0.0f;
  __syncthreads();

  STAGE_S_TILE();
  __syncthreads();

  const v8f vzero = {0.f, 0.f, 0.f, 0.f, 0.f, 0.f, 0.f, 0.f};
  v8f acc[2][NTIL];
#pragma unroll
  for (int mi = 0; mi < 2; ++mi)
#pragma unroll
    for (int nt = 0; nt < NTIL; ++nt) acc[mi][nt] = vzero;

  GEMM_TILE_BODY(acc);

  // per-lane: 8 rows per accumulator = 8 channels, reduce over N-tiles (pixels)
#pragma unroll
  for (int mi = 0; mi < 2; ++mi) {
    const int chBase = (wave * 2 + mi) * 16 + ((lane >> 4) * 8);
#pragma unroll
    for (int r = 0; r < 8; ++r) {
      float s = 0.0f, q = 0.0f;
#pragma unroll
      for (int nt = 0; nt < NTIL; ++nt) {
        float v = acc[mi][nt][r];
        s += v;
        q += v * v;
      }
      atomicAdd(&chSum[chBase + r], s);
      atomicAdd(&chSq[chBase + r], q);
    }
  }
  __syncthreads();
  atomicAdd(&gChSum[tid], chSum[tid]);
  atomicAdd(&gChSq[tid], chSq[tid]);
}

// ---------------- K2: fold BN into scale/shift ----------------
__global__ void k2_bnfinal(const float* __restrict__ chSum,
                           const float* __restrict__ chSq,
                           const float* __restrict__ gamma,
                           const float* __restrict__ beta,
                           float* __restrict__ scale,
                           float* __restrict__ shift) {
  int c = threadIdx.x;
  float inv = 1.0f / NTOT;
  float mean = chSum[c] * inv;
  float var = fmaxf(chSq[c] * inv - mean * mean, 0.0f);
  float sc = gamma[c] * rsqrtf(var + 1e-5f);
  scale[c] = sc;
  shift[c] = beta[c] - mean * sc;
}

// ---------------- K3: WMMA GEMM pass 2 fused: BN+ReLU, L2 norm, MSE, s class stats ----
__global__ __launch_bounds__(256) void k3_main(
    const float* __restrict__ s_feat, const float* __restrict__ t_feat,
    const int* __restrict__ gt, const _Float16* __restrict__ Wh,
    const float* __restrict__ bnScale, const float* __restrict__ bnShift,
    float* __restrict__ gClsSumS, float* __restrict__ gClsSqS,
    float* __restrict__ gCnt, float* __restrict__ gMse) {
  __shared__ _Float16 sStage[PIXB * CPAD];      // ~42 KB
  __shared__ float sOut[COUT * SOPAD];          // ~66.5 KB
  __shared__ float clsSum[NCLS * COUT];         // 21.5 KB
  __shared__ float clsSq[NCLS * COUT];          // 21.5 KB
  __shared__ float scaleSh[COUT];
  __shared__ float shiftSh[COUT];
  __shared__ float normS[PIXB], invS[PIXB], normT[PIXB], invT[PIXB];
  __shared__ float cntSh[NCLS];
  __shared__ int gtT[PIXB];
  __shared__ float red[256];

  const int tid = threadIdx.x;
  const int wave = tid >> 5, lane = tid & 31;
  const int b = blockIdx.x >> 6;          // 64 blocks per batch
  const int tb = blockIdx.x & 63;

  for (int i = tid; i < NCLS * COUT; i += 256) { clsSum[i] = 0.f; clsSq[i] = 0.f; }
  if (tid < NCLS) cntSh[tid] = 0.0f;
  scaleSh[tid] = bnScale[tid];
  shiftSh[tid] = bnShift[tid];
  float msePart = 0.0f;
  const v8f vzero = {0.f, 0.f, 0.f, 0.f, 0.f, 0.f, 0.f, 0.f};

  for (int it = 0; it < 4; ++it) {        // 256 tiles per batch / 64 blocks
    const int p0 = (tb + it * 64) * PIXB;
    __syncthreads();                      // previous iteration fully consumed
    if (tid < PIXB) {
      normS[tid] = 0.0f;
      normT[tid] = 0.0f;
      gtT[tid] = gt[b * P_ + p0 + tid];
    }
    STAGE_S_TILE();
    __syncthreads();

    v8f acc[2][NTIL];
#pragma unroll
    for (int mi = 0; mi < 2; ++mi)
#pragma unroll
      for (int nt = 0; nt < NTIL; ++nt) acc[mi][nt] = vzero;

    GEMM_TILE_BODY(acc);

    // BN affine + ReLU -> sOut, accumulate per-pixel ||s||^2
#pragma unroll
    for (int nt = 0; nt < NTIL; ++nt) {
      const int pix = nt * 16 + (lane & 15);
      float pp = 0.0f;
#pragma unroll
      for (int mi = 0; mi < 2; ++mi) {
        const int chBase = (wave * 2 + mi) * 16 + ((lane >> 4) * 8);
#pragma unroll
        for (int r = 0; r < 8; ++r) {
          const int ch = chBase + r;
          float v = fmaxf(scaleSh[ch] * acc[mi][nt][r] + shiftSh[ch], 0.0f);
          sOut[ch * SOPAD + pix] = v;
          pp += v * v;
        }
      }
      atomicAdd(&normS[pix], pp);
    }

    // t pass 1: ||t||^2 per pixel (coalesced stream)
    const float* tB = t_feat + (size_t)b * COUT * P_ + p0;
#pragma unroll 4
    for (int i = 0; i < (COUT * PIXB) / (256 * 4); ++i) {
      int g = tid + i * 256;
      int c = g >> 4;
      int p = (g & 15) * 4;
      const float4 tv = *(const float4*)(tB + (size_t)c * P_ + p);
      atomicAdd(&normT[p + 0], tv.x * tv.x);
      atomicAdd(&normT[p + 1], tv.y * tv.y);
      atomicAdd(&normT[p + 2], tv.z * tv.z);
      atomicAdd(&normT[p + 3], tv.w * tv.w);
    }
    __syncthreads();
    if (tid < PIXB) {
      invS[tid] = 1.0f / fmaxf(sqrtf(normS[tid]), 1e-12f);
      invT[tid] = 1.0f / fmaxf(sqrtf(normT[tid]), 1e-12f);
    }
    __syncthreads();

    // t pass 2: MSE between unit vectors
#pragma unroll 4
    for (int i = 0; i < (COUT * PIXB) / (256 * 4); ++i) {
      int g = tid + i * 256;
      int c = g >> 4;
      int p = (g & 15) * 4;
      const float4 tv = *(const float4*)(tB + (size_t)c * P_ + p);
      float d;
      d = sOut[c * SOPAD + p + 0] * invS[p + 0] - tv.x * invT[p + 0]; msePart += d * d;
      d = sOut[c * SOPAD + p + 1] * invS[p + 1] - tv.y * invT[p + 1]; msePart += d * d;
      d = sOut[c * SOPAD + p + 2] * invS[p + 2] - tv.z * invT[p + 2]; msePart += d * d;
      d = sOut[c * SOPAD + p + 3] * invS[p + 3] - tv.w * invT[p + 3]; msePart += d * d;
    }

    // pixel-serial class accumulation: thread == channel, no atomics, no races
    for (int p = 0; p < PIXB; ++p) {
      int k = gtT[p];
      k = (k < 0) ? 0 : ((k > NCLS - 1) ? NCLS - 1 : k);
      float sv = sOut[tid * SOPAD + p] * invS[p];
      clsSum[k * COUT + tid] += sv;
      clsSq[k * COUT + tid] += sv * sv;
      if (tid == 0) cntSh[k] += 1.0f;
    }
  }
  __syncthreads();

  // flush class statistics + counts + mse
  for (int k = 0; k < NCLS; ++k) {
    atomicAdd(&gClsSumS[((size_t)b * NCLS + k) * COUT + tid], clsSum[k * COUT + tid]);
    atomicAdd(&gClsSqS[((size_t)b * NCLS + k) * COUT + tid], clsSq[k * COUT + tid]);
  }
  if (tid < NCLS) atomicAdd(&gCnt[b * NCLS + tid], cntSh[tid]);
  red[tid] = msePart;
  __syncthreads();
  for (int s = 128; s > 0; s >>= 1) {
    if (tid < s) red[tid] += red[tid + s];
    __syncthreads();
  }
  if (tid == 0) atomicAdd(gMse, red[0]);
}

// ---------------- Kt: class stats of normalized t (GEMM-free) ----------------
__global__ __launch_bounds__(256) void kt_tstats(
    const float* __restrict__ t_feat, const int* __restrict__ gt,
    float* __restrict__ gClsSumT, float* __restrict__ gClsSqT) {
  __shared__ float tOut[COUT * SOPAD];
  __shared__ float clsSum[NCLS * COUT];
  __shared__ float clsSq[NCLS * COUT];
  __shared__ float normT[PIXB], invT[PIXB];
  __shared__ int gtT[PIXB];
  const int tid = threadIdx.x;
  const int b = blockIdx.x >> 6;
  const int tb = blockIdx.x & 63;
  for (int i = tid; i < NCLS * COUT; i += 256) { clsSum[i] = 0.f; clsSq[i] = 0.f; }

  for (int it = 0; it < 4; ++it) {
    const int p0 = (tb + it * 64) * PIXB;
    __syncthreads();
    if (tid < PIXB) {
      normT[tid] = 0.0f;
      gtT[tid] = gt[b * P_ + p0 + tid];
    }
    __syncthreads();
    const float* tB = t_feat + (size_t)b * COUT * P_ + p0;
#pragma unroll 4
    for (int i = 0; i < (COUT * PIXB) / (256 * 4); ++i) {
      int g = tid + i * 256;
      int c = g >> 4;
      int p = (g & 15) * 4;
      const float4 tv = *(const float4*)(tB + (size_t)c * P_ + p);
      tOut[c * SOPAD + p + 0] = tv.x;
      tOut[c * SOPAD + p + 1] = tv.y;
      tOut[c * SOPAD + p + 2] = tv.z;
      tOut[c * SOPAD + p + 3] = tv.w;
      atomicAdd(&normT[p + 0], tv.x * tv.x);
      atomicAdd(&normT[p + 1], tv.y * tv.y);
      atomicAdd(&normT[p + 2], tv.z * tv.z);
      atomicAdd(&normT[p + 3], tv.w * tv.w);
    }
    __syncthreads();
    if (tid < PIXB) invT[tid] = 1.0f / fmaxf(sqrtf(normT[tid]), 1e-12f);
    __syncthreads();
    for (int p = 0; p < PIXB; ++p) {
      int k = gtT[p];
      k = (k < 0) ? 0 : ((k > NCLS - 1) ? NCLS - 1 : k);
      float tv = tOut[tid * SOPAD + p] * invT[p];
      clsSum[k * COUT + tid] += tv;
      clsSq[k * COUT + tid] += tv * tv;
    }
  }
  __syncthreads();
  for (int k = 0; k < NCLS; ++k) {
    atomicAdd(&gClsSumT[((size_t)b * NCLS + k) * COUT + tid], clsSum[k * COUT + tid]);
    atomicAdd(&gClsSqT[((size_t)b * NCLS + k) * COUT + tid], clsSq[k * COUT + tid]);
  }
}

// ---------------- K5: final scalar reduction ----------------
__global__ __launch_bounds__(256) void k5_final(
    const float* __restrict__ sumS, const float* __restrict__ sqS,
    const float* __restrict__ sumT, const float* __restrict__ sqT,
    const float* __restrict__ cnt, const float* __restrict__ mseAcc,
    float* __restrict__ out) {
  __shared__ float redL[256];
  __shared__ float redC[256];
  int tid = threadIdx.x;
  float accL = 0.0f, accC = 0.0f;
  for (int pair = tid; pair < B_ * NCLS; pair += 256) {
    int k = pair % NCLS;
    float c = cnt[pair];
    if (k != 0 && c > 1.0f) {
      float cs = fmaxf(c, 1.0f);
      float dn = fmaxf(c - 1.0f, 1.0f);
      const float* ss = sumS + (size_t)pair * COUT;
      const float* qs = sqS + (size_t)pair * COUT;
      const float* st = sumT + (size_t)pair * COUT;
      const float* qt = sqT + (size_t)pair * COUT;
      float lm = 0.0f, lv = 0.0f;
      for (int ch = 0; ch < COUT; ++ch) {
        float mus = ss[ch] / cs;
        float mut = st[ch] / cs;
        float vs = (qs[ch] - c * mus * mus) / dn + 1e-6f;
        float vt = (qt[ch] - c * mut * mut) / dn + 1e-6f;
        float dm = mus - mut;
        float dv = vs - vt;
        lm += dm * dm;
        lv += dv * dv;
      }
      accL += 0.5f * (lm / (float)COUT) + 0.5f * (lv / (float)COUT);
      accC += 1.0f;
    }
  }
  redL[tid] = accL;
  redC[tid] = accC;
  __syncthreads();
  for (int s = 128; s > 0; s >>= 1) {
    if (tid < s) { redL[tid] += redL[tid + s]; redC[tid] += redC[tid + s]; }
    __syncthreads();
  }
  if (tid == 0) {
    float mse = mseAcc[0] / ((float)B_ * (float)COUT * (float)P_);
    out[0] = redL[0] / (redC[0] + 1e-7f) + 0.5f * mse;
  }
}

// ---------------- launch ----------------
extern "C" void kernel_launch(void* const* d_in, const int* in_sizes, int n_in,
                              void* d_out, int out_size, void* d_ws, size_t ws_size,
                              hipStream_t stream) {
  const float* s_feat = (const float*)d_in[0];
  const float* t_feat = (const float*)d_in[1];
  const int* gt = (const int*)d_in[2];
  const float* W_proj = (const float*)d_in[3];
  const float* gamma = (const float*)d_in[4];
  const float* beta = (const float*)d_in[5];

  float* ws = (float*)d_ws;
  _Float16* Wh = (_Float16*)(ws + WS_WH);

  k_zero<<<(WS_ZEROCNT + 255) / 256, 256, 0, stream>>>(ws, WS_ZEROCNT);
  k_wcvt<<<(COUT * CIN) / 256, 256, 0, stream>>>(W_proj, Wh);
  k1_bnstats<<<2048, 256, 0, stream>>>(s_feat, Wh, ws + WS_CHSUM, ws + WS_CHSQ);
  k2_bnfinal<<<1, 256, 0, stream>>>(ws + WS_CHSUM, ws + WS_CHSQ, gamma, beta,
                                    ws + WS_SCALE, ws + WS_SHIFT);
  k3_main<<<512, 256, 0, stream>>>(s_feat, t_feat, gt, Wh,
                                   ws + WS_SCALE, ws + WS_SHIFT,
                                   ws + WS_CLS_SS, ws + WS_CLS_QS,
                                   ws + WS_CNT, ws + WS_MSE);
  kt_tstats<<<512, 256, 0, stream>>>(t_feat, gt, ws + WS_CLS_ST, ws + WS_CLS_QT);
  k5_final<<<1, 256, 0, stream>>>(ws + WS_CLS_SS, ws + WS_CLS_QS,
                                  ws + WS_CLS_ST, ws + WS_CLS_QT,
                                  ws + WS_CNT, ws + WS_MSE, (float*)d_out);
}
